// GATModel2_13804024889636
// MI455X (gfx1250) — matrix operand order repaired and checked by reading, not compiled, not run on hardware
//
#include <hip/hip_runtime.h>

#define IN_F   128
#define HID    32
#define HEADS  8
#define C1     (HEADS * HID)   /* 256 */
#define OUT_F  64
#define NEG_SLOPE 0.2f

typedef __attribute__((ext_vector_type(16))) _Float16 v16h;
typedef __attribute__((ext_vector_type(8)))  _Float16 v8h;
typedef __attribute__((ext_vector_type(8)))  float    v8f;

__device__ __forceinline__ unsigned short f32_to_f16bits(float f) {
  _Float16 h = (_Float16)f;
  return __builtin_bit_cast(unsigned short, h);
}

// ---------------------------------------------------------------- conversions
__global__ void k_f32_to_f16(const float* __restrict__ in,
                             unsigned short* __restrict__ out, long n) {
  long i = (long)blockIdx.x * blockDim.x + threadIdx.x;
  if (i < n) out[i] = f32_to_f16bits(in[i]);
}

// W[K][C] (f32, row-major) -> Wt[C][K] (f16) so that B columns are contiguous
__global__ void k_transpose_f16(const float* __restrict__ W,
                                unsigned short* __restrict__ Wt, int K, int C) {
  int i = blockIdx.x * blockDim.x + threadIdx.x;
  if (i >= C * K) return;
  int c = i / K, k = i % K;
  Wt[i] = f32_to_f16bits(W[(long)k * C + c]);
}

// ------------------------------------------------------------------ WMMA GEMM
// C[M,Ncols] = A[M,K] * B[K,Ncols], A f16 row-major, B given transposed
// (Bt[Ncols][K], f16). One wave computes one 16x16 tile, K-loop in steps of 32.
__global__ void k_gemm_wmma_f16(const unsigned short* __restrict__ A,
                                const unsigned short* __restrict__ Bt,
                                float* __restrict__ C,
                                int M, int Ncols, int K) {
  const int lane   = threadIdx.x & 31;
  const int wave   = threadIdx.x >> 5;
  const int nTiles = Ncols >> 4;
  const int tile   = blockIdx.x * (blockDim.x >> 5) + wave;
  if (tile >= (M >> 4) * nTiles) return;          // wave-uniform: EXEC stays all-1s
  const int mt = tile / nTiles, nt = tile % nTiles;
  const int r  = lane & 15;                       // row in A-tile / col in B-tile
  const int kg = lane >> 4;                       // lane-group selects K half
  const unsigned short* arow = A  + (long)((mt << 4) + r) * K;
  const unsigned short* brow = Bt + (long)((nt << 4) + r) * K;
  v8f acc = {};
  for (int kb = 0; kb < K; kb += 32) {
    const v8h* pa = (const v8h*)(arow + kb);
    const v8h* pb = (const v8h*)(brow + kb);
    // A 16x32 f16 layout: lanes 0-15 hold K = kb+{0..7,16..23}, lanes 16-31
    // hold K = kb+{8..15,24..31}  (ISA 05_wmma 16-bit A table)
    v16h a = __builtin_shufflevector(pa[kg], pa[2 + kg],
        0,1,2,3,4,5,6,7,8,9,10,11,12,13,14,15);
    // B 32x16 f16 layout: same K order as A: lanes 0-15 hold K = kb+{0..7,16..23},
    // lanes 16-31 hold K = kb+{8..15,24..31}
    v16h b = __builtin_shufflevector(pb[kg], pb[2 + kg],
        0,1,2,3,4,5,6,7,8,9,10,11,12,13,14,15);
    acc = __builtin_amdgcn_wmma_f32_16x16x32_f16(false, a, false, b,
                                                 (short)0, acc, false, false);
  }
  // D layout: VGPR v, lane l -> (M = v + 8*(l>>4), N = l&15)
  float* crow = C + (long)(mt << 4) * Ncols + (nt << 4);
#pragma unroll
  for (int v = 0; v < 8; v++)
    crow[(long)(v + 8 * kg) * Ncols + r] = acc[v];
}

// --------------------------------------------------- per-node attention terms
// el[n,h] = sum_f feat[n,h,f]*al[h,f]  ;  er likewise
__global__ void k_attn_scores(const float* __restrict__ feat,
                              const float* __restrict__ al,
                              const float* __restrict__ ar,
                              float* __restrict__ el, float* __restrict__ er,
                              int N, int H, int F) {
  int idx = blockIdx.x * blockDim.x + threadIdx.x;   // n*H + h
  if (idx >= N * H) return;
  int h = idx % H, n = idx / H;
  const float* f  = feat + (long)n * H * F + (long)h * F;
  const float* alh = al + (long)h * F;
  const float* arh = ar + (long)h * F;
  float sl = 0.f, sr = 0.f;
  for (int i = 0; i < F; i++) { sl += f[i] * alh[i]; sr += f[i] * arh[i]; }
  el[idx] = sl; er[idx] = sr;
}

// -------------------------------------- edge softmax numerators + denominator
__global__ void k_edge_softmax(const float* __restrict__ el,
                               const float* __restrict__ er,
                               const int* __restrict__ src,
                               const int* __restrict__ dst,
                               float* __restrict__ ee, float* __restrict__ den,
                               int E, int H) {
  int idx = blockIdx.x * blockDim.x + threadIdx.x;   // e*H + h
  if (idx >= E * H) return;
  int h = idx % H, e = idx / H;
  int s = src[e], d = dst[e];
  float v = el[(long)s * H + h] + er[(long)d * H + h];
  v = v > 0.f ? v : NEG_SLOPE * v;                   // leaky_relu
  float ex = __expf(v);                              // scores ~O(1): no max shift needed
  ee[idx] = ex;
  atomicAdd(&den[(long)d * H + h], ex);
}

// ------------------------------------------- gather-scale-scatter aggregation
// agg[dst, h, f] += (ee/den[dst]) * feat[src, h, f];  one thread per (edge,col)
__global__ void k_edge_aggregate(const float* __restrict__ feat,
                                 const float* __restrict__ ee,
                                 const float* __restrict__ den,
                                 const int* __restrict__ src,
                                 const int* __restrict__ dst,
                                 float* __restrict__ agg,
                                 int E, int H, int F) {
  int cols = H * F;
  int epb  = blockDim.x / cols;
  int le   = threadIdx.x / cols;
  int c    = threadIdx.x % cols;
  int e    = blockIdx.x * epb + le;
  if (e >= E) return;
  int s = src[e], d = dst[e];
  int h = c / F;
  float alpha = ee[(long)e * H + h] / den[(long)d * H + h];
  atomicAdd(&agg[(long)d * cols + c], alpha * feat[(long)s * cols + c]);
}

// ----------------------------------- bias + relu + convert to f16 for GEMM2 A
__global__ void k_bias_relu_f16(const float* __restrict__ in,
                                const float* __restrict__ bias,
                                unsigned short* __restrict__ out,
                                long n, int cols) {
  long i = (long)blockIdx.x * blockDim.x + threadIdx.x;
  if (i >= n) return;
  float v = in[i] + bias[i % cols];
  out[i] = f32_to_f16bits(v > 0.f ? v : 0.f);
}

// ---------------------------------------- final: relu, linear head, sigmoid
__global__ void k_final(const float* __restrict__ agg2,
                        const float* __restrict__ b2,
                        const float* __restrict__ pW, const float* __restrict__ pb,
                        float* __restrict__ out, int N) {
  int n = blockIdx.x * blockDim.x + threadIdx.x;
  if (n >= N) return;
  const float* row = agg2 + (long)n * OUT_F;
  float s = 0.f;
  for (int f = 0; f < OUT_F; f++) {
    float v = row[f] + b2[f];
    v = v > 0.f ? v : 0.f;
    s += v * pW[f];
  }
  s += pb[0];
  out[n] = 1.f / (1.f + __expf(-s));
}

// =============================================================== host driver
extern "C" void kernel_launch(void* const* d_in, const int* in_sizes, int n_in,
                              void* d_out, int out_size, void* d_ws, size_t ws_size,
                              hipStream_t stream) {
  const float* features = (const float*)d_in[0];
  const int*   src      = (const int*)  d_in[1];
  const int*   dst      = (const int*)  d_in[2];
  /* d_in[3] edge_types: unused by the reference */
  const float* W1  = (const float*)d_in[4];
  const float* al1 = (const float*)d_in[5];
  const float* ar1 = (const float*)d_in[6];
  const float* b1  = (const float*)d_in[7];
  const float* W2  = (const float*)d_in[8];
  const float* al2 = (const float*)d_in[9];
  const float* ar2 = (const float*)d_in[10];
  const float* b2  = (const float*)d_in[11];
  const float* pW  = (const float*)d_in[12];
  const float* pb  = (const float*)d_in[13];
  const int N = in_sizes[0] / IN_F;
  const int E = in_sizes[1];
  float* out = (float*)d_out;

  // ---- workspace layout (256B aligned, with arena reuse)
  char*  ws  = (char*)d_ws;
  size_t off = 0;
  auto alloc = [&](size_t bytes) -> void* {
    void* p = ws + off;
    off = (off + bytes + 255) & ~(size_t)255;
    return p;
  };
  unsigned short* W1t   = (unsigned short*)alloc((size_t)C1 * IN_F * 2);
  unsigned short* W2t   = (unsigned short*)alloc((size_t)OUT_F * C1 * 2);
  float* el1  = (float*)alloc((size_t)N * HEADS * 4);
  float* er1  = (float*)alloc((size_t)N * HEADS * 4);
  float* ee1  = (float*)alloc((size_t)E * HEADS * 4);
  float* den1 = (float*)alloc((size_t)N * HEADS * 4);
  // arena A: f16 features for GEMM1, later reused as f16 x for GEMM2
  unsigned short* featB = (unsigned short*)alloc((size_t)N * C1 * 2);
  unsigned short* xB    = featB;
  // arena B: feat1 (f32), later reused for ALL conv-2 buffers (58.8MB < 102.4MB)
  float* feat1 = (float*)alloc((size_t)N * C1 * 4);
  float* agg1  = (float*)alloc((size_t)N * C1 * 4);
  {
    char* arena = (char*)feat1;
    // laid out after feat1 is dead (post edge_aggregate for layer 1)
    (void)arena;
  }
  float* feat2 = (float*)((char*)feat1);
  float* el2   = (float*)((char*)feat1 + (size_t)N * OUT_F * 4);
  float* er2   = el2 + N;
  float* ee2   = er2 + N;
  float* den2  = ee2 + E;
  float* agg2  = den2 + N;

  // ---- 0. convert operands to f16 (features, W1^T, W2^T)
  long nfeat = (long)N * IN_F;
  k_f32_to_f16 <<<(unsigned)((nfeat + 255) / 256), 256, 0, stream>>>(features, featB, nfeat);
  k_transpose_f16<<<(C1 * IN_F + 255) / 256, 256, 0, stream>>>(W1, W1t, IN_F, C1);
  k_transpose_f16<<<(OUT_F * C1 + 255) / 256, 256, 0, stream>>>(W2, W2t, C1, OUT_F);

  // ---- 1. GEMM1: feat1[N,256] = features @ W1   (WMMA f16 -> f32)
  {
    int tiles = (N / 16) * (C1 / 16);
    k_gemm_wmma_f16<<<(tiles + 3) / 4, 128, 0, stream>>>(featB, W1t, feat1, N, C1, IN_F);
  }

  // ---- 2. attention scores, edge softmax, aggregation (layer 1)
  k_attn_scores<<<(N * HEADS + 255) / 256, 256, 0, stream>>>(feat1, al1, ar1, el1, er1, N, HEADS, HID);
  hipMemsetAsync(den1, 0, (size_t)N * HEADS * 4, stream);
  k_edge_softmax<<<(E * HEADS + 255) / 256, 256, 0, stream>>>(el1, er1, src, dst, ee1, den1, E, HEADS);
  hipMemsetAsync(agg1, 0, (size_t)N * C1 * 4, stream);
  k_edge_aggregate<<<E, 256, 0, stream>>>(feat1, ee1, den1, src, dst, agg1, E, HEADS, HID);

  // ---- 3. x = relu(agg1 + b1), converted to f16 (feat1 arena now free)
  k_bias_relu_f16<<<(unsigned)(((long)N * C1 + 255) / 256), 256, 0, stream>>>(agg1, b1, xB, (long)N * C1, C1);

  // ---- 4. GEMM2: feat2[N,64] = x @ W2
  {
    int tiles = (N / 16) * (OUT_F / 16);
    k_gemm_wmma_f16<<<(tiles + 3) / 4, 128, 0, stream>>>(xB, W2t, feat2, N, OUT_F, C1);
  }

  // ---- 5. layer-2 attention (single head)
  k_attn_scores<<<(N + 255) / 256, 256, 0, stream>>>(feat2, al2, ar2, el2, er2, N, 1, OUT_F);
  hipMemsetAsync(den2, 0, (size_t)N * 4, stream);
  k_edge_softmax<<<(E + 255) / 256, 256, 0, stream>>>(el2, er2, src, dst, ee2, den2, E, 1);
  hipMemsetAsync(agg2, 0, (size_t)N * OUT_F * 4, stream);
  k_edge_aggregate<<<(E + 3) / 4, 256, 0, stream>>>(feat2, ee2, den2, src, dst, agg2, E, 1, OUT_F);

  // ---- 6. head: relu, dot with pW, sigmoid
  k_final<<<(N + 255) / 256, 256, 0, stream>>>(agg2, b2, pW, pb, out, N);
}
